// DPI_7438883357622
// MI455X (gfx1250) — compile-verified, matrix operand order checked by reading, not run on hardware
//
#include <hip/hip_runtime.h>
#include <hip/hip_bf16.h>

typedef __attribute__((ext_vector_type(16))) _Float16 v16h;
typedef __attribute__((ext_vector_type(8)))  _Float16 v8h;
typedef __attribute__((ext_vector_type(8)))  float    v8f;

#define HDIM    128
#define TM      128         // edges per block
#define BLOCK   512         // 16 waves (wave32)
#define XPITCH  264         // halves per X row (256 + 8 pad) -> 528B, 16B aligned
#define H1PITCH 520         // halves per H1 row (512 + 8 pad) -> 1040B, 16B aligned
#define LDS_X_BYTES   (TM * XPITCH * 2)    // 67584
#define LDS_H1_BYTES  (TM * H1PITCH * 2)   // 133120 (total ~196KB of 320KB/WGP)

// Build a 16-half WMMA fragment from two contiguous 16B chunks at p and p+16
// (CDNA5 16-bit A/B per-lane layout: halves K+0..7 then K+16..23, lane>=16 gets +8).
__device__ __forceinline__ v16h frag16(const _Float16* p) {
  v8h lo = *(const v8h*)p;
  v8h hi = *(const v8h*)(p + 16);
  v16h r;
#pragma unroll
  for (int i = 0; i < 8; ++i) { r[i] = lo[i]; r[i + 8] = hi[i]; }
  return r;
}

__device__ __forceinline__ v8f wmma16(v16h a, v16h b, v8f c) {
  return __builtin_amdgcn_wmma_f32_16x16x32_f16(false, a, false, b, (short)0, c,
                                                false, false);
}

// Pre-pass: W1 [256,512] f32 -> W1t f16 [512][256]; W2 [512,128] f32 -> W2t f16 [128][512]
__global__ void convert_weights_kernel(const float* __restrict__ W1,
                                       const float* __restrict__ W2,
                                       _Float16* __restrict__ W1t,
                                       _Float16* __restrict__ W2t) {
  int idx = blockIdx.x * blockDim.x + threadIdx.x;
  if (idx < 256 * 512) {
    int k = idx >> 9, n = idx & 511;            // W1[k][n]
    W1t[n * 256 + k] = (_Float16)W1[idx];
  }
  if (idx < 512 * 128) {
    int k = idx >> 7, n = idx & 127;            // W2[k][n]
    W2t[n * 512 + k] = (_Float16)W2[idx];
  }
}

__global__ __launch_bounds__(BLOCK)
void edge_mlp_kernel(const float* __restrict__ emb,
                     const int*   __restrict__ eidx,    // [2, E]
                     const _Float16* __restrict__ W1t,  // [512][256] f16
                     const float* __restrict__ b1,      // [512]
                     const _Float16* __restrict__ W2t,  // [128][512] f16
                     const float* __restrict__ b2,      // [128]
                     const float* __restrict__ W3,      // [128]
                     const float* __restrict__ b3,      // [1]
                     float* __restrict__ out,           // [E]
                     int E, int nNodes) {
  __shared__ __align__(16) unsigned char lds[LDS_X_BYTES + LDS_H1_BYTES];
  _Float16* Xs  = (_Float16*)lds;                   // [TM][XPITCH] f16
  _Float16* H1s = (_Float16*)(lds + LDS_X_BYTES);   // [TM][H1PITCH] f16
  float*    H2s = (float*)lds;                      // [TM][128] f32, overlays Xs

  const int t    = threadIdx.x;
  const int wave = t >> 5;
  const int lane = t & 31;
  const int lrow = lane & 15;
  const int koff  = (lane >> 4) << 3;  // +8 halves for upper half-wave
  const int rbase = (lane >> 4) << 3;  // C/D row offset for upper half-wave
  const int mg = wave >> 3;            // M-group: rows [mg*64, mg*64+64)
  const int ng = wave & 7;             // N-group

  // ---------------- Stage 0: gather + f32->f16 into X tile -----------------
  {
    int e   = t >> 2;                 // 0..127
    int seg = t & 3;                  // 4 segs x 64 floats = 256-wide row
    int eg  = blockIdx.x * TM + e;
    int egc = eg < E ? eg : (E - 1);  // clamp tail so EXEC stays full for WMMA
    int nc  = eidx[egc];
    int nr  = eidx[E + egc];
    nc = (nc < 0) ? 0 : (nc >= nNodes ? nNodes - 1 : nc);
    nr = (nr < 0) ? 0 : (nr >= nNodes ? nNodes - 1 : nr);
    const float* src = (seg < 2) ? (emb + (size_t)nc * HDIM + seg * 64)
                                 : (emb + (size_t)nr * HDIM + (seg - 2) * 64);
    _Float16* dst = Xs + e * XPITCH + seg * 64;
#pragma unroll 4
    for (int i = 0; i < 64; i += 8) {
      float4 u = *(const float4*)(src + i);
      float4 v = *(const float4*)(src + i + 4);
      v8h h;
      h[0] = (_Float16)u.x; h[1] = (_Float16)u.y;
      h[2] = (_Float16)u.z; h[3] = (_Float16)u.w;
      h[4] = (_Float16)v.x; h[5] = (_Float16)v.y;
      h[6] = (_Float16)v.z; h[7] = (_Float16)v.w;
      *(v8h*)(dst + i) = h;
    }
  }
  __syncthreads();

  // ------- Layer 1: H1 = relu(X @ W1 + b1)  [128,256]x[256,512] ------------
  // wave grid: 2 M-groups (64 rows) x 8 N-groups (64 cols); 4x4 tiles per wave
  {
    const int m0base = mg * 64;
    const int n0base = ng * 64;
    v8f zero = {0.f, 0.f, 0.f, 0.f, 0.f, 0.f, 0.f, 0.f};
    v8f acc[4][4];
#pragma unroll
    for (int i = 0; i < 4; ++i)
#pragma unroll
      for (int j = 0; j < 4; ++j) acc[i][j] = zero;

    // prefetch our W2t slice for layer 2 while doing layer 1
    __builtin_prefetch(W2t + (size_t)(ng * 16 + lrow) * 512, 0, 3);

    // unroll 1: keep live ranges small -- 16 waves/WG caps us at ~256 VGPRs/wave
#pragma unroll 1
    for (int kt = 0; kt < 8; ++kt) {  // K = 256, 32 per WMMA
      int k0 = kt * 32;
      v16h a[4];
#pragma unroll
      for (int mt = 0; mt < 4; ++mt)
        a[mt] = frag16(Xs + (size_t)(m0base + mt * 16 + lrow) * XPITCH + k0 + koff);
#pragma unroll
      for (int nt = 0; nt < 4; ++nt) {
        v16h b = frag16(W1t + (size_t)(n0base + nt * 16 + lrow) * 256 + k0 + koff);
#pragma unroll
        for (int mt = 0; mt < 4; ++mt)
          acc[mt][nt] = wmma16(a[mt], b, acc[mt][nt]);
      }
    }
    // epilogue: bias + relu, write f16 H1 tile to LDS
#pragma unroll
    for (int nt = 0; nt < 4; ++nt) {
      int n = n0base + nt * 16 + lrow;
      float bias = b1[n];
#pragma unroll
      for (int mt = 0; mt < 4; ++mt) {
#pragma unroll
        for (int r = 0; r < 8; ++r) {
          float v = acc[mt][nt][r] + bias;
          v = v > 0.f ? v : 0.f;
          H1s[(size_t)(m0base + mt * 16 + rbase + r) * H1PITCH + n] = (_Float16)v;
        }
      }
    }
  }
  __syncthreads();   // Xs dead after this point; H2s may overlay it

  // ------- Layer 2: H2 = relu(H1 @ W2 + b2)  [128,512]x[512,128] -----------
  // wave grid: 2 M-groups (64 rows) x 8 N-groups (16 cols); 4x1 tiles per wave
  {
    const int m0base = mg * 64;
    const int n0 = ng * 16;
    v8f zero = {0.f, 0.f, 0.f, 0.f, 0.f, 0.f, 0.f, 0.f};
    v8f acc2[4] = {zero, zero, zero, zero};
#pragma unroll 2
    for (int kt = 0; kt < 16; ++kt) { // K = 512
      int k0 = kt * 32;
      v16h b = frag16(W2t + (size_t)(n0 + lrow) * 512 + k0 + koff);
#pragma unroll
      for (int mt = 0; mt < 4; ++mt) {
        v16h a = frag16(H1s + (size_t)(m0base + mt * 16 + lrow) * H1PITCH + k0 + koff);
        acc2[mt] = wmma16(a, b, acc2[mt]);
      }
    }
    int n = n0 + lrow;
    float bias = b2[n];
#pragma unroll
    for (int mt = 0; mt < 4; ++mt) {
#pragma unroll
      for (int r = 0; r < 8; ++r) {
        float v = acc2[mt][r] + bias;
        v = v > 0.f ? v : 0.f;
        H2s[(size_t)(m0base + mt * 16 + rbase + r) * HDIM + n] = v;
      }
    }
  }
  __syncthreads();

  // ---------------- Layer 3: sigmoid(H2 @ W3 + b3), one thread per edge ----
  if (t < TM) {
    const float* hr = H2s + (size_t)t * HDIM;
    float s = b3[0];
#pragma unroll 4
    for (int i = 0; i < HDIM; ++i) s += hr[i] * W3[i];
    int eg = blockIdx.x * TM + t;
    if (eg < E) out[eg] = 1.0f / (1.0f + __expf(-s));
  }
}

extern "C" void kernel_launch(void* const* d_in, const int* in_sizes, int n_in,
                              void* d_out, int out_size, void* d_ws, size_t ws_size,
                              hipStream_t stream) {
  const float* emb  = (const float*)d_in[0];
  const int*   eidx = (const int*)d_in[1];
  const float* W1   = (const float*)d_in[2];
  const float* b1   = (const float*)d_in[3];
  const float* W2   = (const float*)d_in[4];
  const float* b2   = (const float*)d_in[5];
  const float* W3   = (const float*)d_in[6];
  const float* b3   = (const float*)d_in[7];
  float* out = (float*)d_out;

  const int E      = in_sizes[1] / 2;        // edge_index is [2, E]
  const int nNodes = in_sizes[0] / HDIM;

  _Float16* W1t = (_Float16*)d_ws;                                   // 512*256*2 = 256 KB
  _Float16* W2t = (_Float16*)((char*)d_ws + (size_t)512 * 256 * 2);  // 128*512*2 = 128 KB

  int convN = 256 * 512;
  convert_weights_kernel<<<(convN + 255) / 256, 256, 0, stream>>>(W1, W2, W1t, W2t);

  int grid = (E + TM - 1) / TM;
  edge_mlp_kernel<<<grid, BLOCK, 0, stream>>>(emb, eidx, W1t, b1, W2t, b2,
                                              W3, b3, out, E, nNodes);
}
